// CrossScaleAttention_43868795961528
// MI455X (gfx1250) — compile-verified, matrix operand order checked by reading, not run on hardware
//
#include <hip/hip_runtime.h>
#include <hip/hip_bf16.h>

typedef __bf16 bf16_t;
typedef __attribute__((ext_vector_type(16))) __bf16 v16bf;
typedef __attribute__((ext_vector_type(8)))  float  v8f;

#define DIM     256
#define NSCALE  3
#define BATCH   4
#define NTOT    5376           // 4096 + 1024 + 256
#define ATT_SCALE 0.17677669529663687f  // (256/8)^-0.5

__constant__ int c_N[3]    = {4096, 1024, 256};
__constant__ int c_pref[3] = {0, 4096, 5120};   // prefix sums of N

// ---------------------------------------------------------------------------
// fp32 -> bf16 elementwise convert (weights)
// ---------------------------------------------------------------------------
__global__ void cvt_kernel(const float* __restrict__ s, bf16_t* __restrict__ d, int n) {
    int i = blockIdx.x * 256 + threadIdx.x;
    if (i < n) d[i] = (bf16_t)s[i];
}

// ---------------------------------------------------------------------------
// pack feats [B,C,H,W] -> X bf16 [scale][b][token][c]
// ---------------------------------------------------------------------------
__global__ void pack_kernel(const float* __restrict__ f0, const float* __restrict__ f1,
                            const float* __restrict__ f2, bf16_t* __restrict__ X) {
    long idx = (long)blockIdx.x * 256 + threadIdx.x;    // [0, BATCH*NTOT*DIM)
    const long c0 = (long)BATCH * 4096 * DIM;
    const long c1 = (long)BATCH * 1024 * DIM;
    long r = idx; const float* f; int Nj;
    if (r < c0)            { f = f0; Nj = 4096; }
    else if (r < c0 + c1)  { f = f1; Nj = 1024; r -= c0; }
    else                   { f = f2; Nj = 256;  r -= c0 + c1; }
    int  c  = (int)(r % DIM);
    long tb = r / DIM;
    int  t  = (int)(tb % Nj);
    int  b  = (int)(tb / Nj);
    X[idx] = (bf16_t)f[((long)(b * DIM + c)) * Nj + t];
}

// ---------------------------------------------------------------------------
// projection GEMM: Y = X @ W^T + bias   (one wave per 16x16 output tile)
// mode 0: Y row-major [b][t][d]   (Q, K)
// mode 1: Y transposed  [b][d][t] (V^T)
// ---------------------------------------------------------------------------
__global__ __launch_bounds__(64) void proj_kernel(
        const bf16_t* __restrict__ X, const bf16_t* __restrict__ W,
        const float* __restrict__ bias, bf16_t* __restrict__ Y, int mode) {
    int wv   = blockIdx.x * 2 + (threadIdx.x >> 5);
    int lane = threadIdx.x & 31;
    int j, r;
    if (wv < 4 * 4096)               { j = 0; r = wv; }
    else if (wv < 4 * 4096 + 4*1024) { j = 1; r = wv - 4 * 4096; }
    else                             { j = 2; r = wv - (4*4096 + 4*1024); }
    int Nj = c_N[j];
    int b  = r / Nj;  int rr = r % Nj;
    int mt = rr >> 4, nt = rr & 15;
    int col = lane & 15, hl = lane >> 4;

    const bf16_t* Arow = X + ((long)c_pref[j] * BATCH + (long)b * Nj + mt*16 + col) * DIM + hl*16;
    const bf16_t* Brow = W + ((long)j * DIM + nt*16 + col) * DIM + hl*16;

    v8f acc = {};
#pragma unroll
    for (int k = 0; k < 8; ++k) {
        v16bf a  = *(const v16bf*)(Arow + k*32);
        v16bf bb = *(const v16bf*)(Brow + k*32);
        acc = __builtin_amdgcn_wmma_f32_16x16x32_bf16(false, a, false, bb, (short)0, acc, false, false);
    }
    int   dcol = nt*16 + col;
    float bs   = bias[j * DIM + dcol];
    if (mode == 0) {
        bf16_t* out = Y + ((long)c_pref[j] * BATCH + (long)b * Nj + mt*16) * DIM;
#pragma unroll
        for (int v = 0; v < 8; ++v) {
            int t = v + 8*hl;
            out[(long)t * DIM + dcol] = (bf16_t)(acc[v] + bs);
        }
    } else {
        bf16_t* out = Y + (long)c_pref[j] * BATCH * DIM;
#pragma unroll
        for (int v = 0; v < 8; ++v) {
            int t = mt*16 + v + 8*hl;
            out[((long)b * DIM + dcol) * Nj + t] = (bf16_t)(acc[v] + bs);
        }
    }
}

// ---------------------------------------------------------------------------
// fused attention + out-proj + residual + layernorm
// one wave handles one (scale i, batch b, 16-query tile)
// ---------------------------------------------------------------------------
__global__ __launch_bounds__(64) void attn_kernel(
        const bf16_t* __restrict__ Q, const bf16_t* __restrict__ K,
        const bf16_t* __restrict__ Vt, const bf16_t* __restrict__ Wo,
        const float* __restrict__ ob, const float* __restrict__ lng,
        const float* __restrict__ lnb,
        const float* __restrict__ f0, const float* __restrict__ f1,
        const float* __restrict__ f2, float* __restrict__ out) {
    __shared__ float outl_s[2][16][DIM];                 // per-wave fp32 combined
    __shared__ __align__(32) bf16_t pl_s[2][16][32];     // per-wave P tile

    int w    = threadIdx.x >> 5;
    int lane = threadIdx.x & 31;
    int col  = lane & 15, hl = lane >> 4;
    int wv   = blockIdx.x * 2 + w;
    int i, r;
    if (wv < 1024)      { i = 0; r = wv; }
    else if (wv < 1280) { i = 1; r = wv - 1024; }
    else                { i = 2; r = wv - 1280; }
    int Ni     = c_N[i];
    int qtiles = Ni >> 4;
    int b      = r / qtiles;
    int qt0    = (r % qtiles) * 16;

    float  (*outl)[DIM] = outl_s[w];
    bf16_t (*pl)[32]    = pl_s[w];

    const bf16_t* Qb = Q + ((long)c_pref[i] * BATCH + (long)b * Ni + qt0 + col) * DIM + hl*16;

    v8f pv[16];
    const v8f vzero = {};

    for (int j = 0; j < NSCALE; ++j) {
        int Nj = c_N[j];
        const bf16_t* Kb = K  + ((long)c_pref[j] * BATCH + (long)b * Nj) * DIM;
        const bf16_t* Vb = Vt + (long)c_pref[j] * BATCH * DIM + (long)b * DIM * Nj;

        float m[8], l[8];
#pragma unroll
        for (int v = 0; v < 8; ++v) { m[v] = -1e30f; l[v] = 0.f; }
#pragma unroll
        for (int nb = 0; nb < 16; ++nb) pv[nb] = vzero;

        for (int kb = 0; kb < Nj; kb += 32) {
            v8f s0 = {}, s1 = {};
            const bf16_t* K0 = Kb + ((long)(kb + col)) * DIM + hl*16;
            const bf16_t* K1 = Kb + ((long)(kb + 16 + col)) * DIM + hl*16;
#pragma unroll
            for (int k = 0; k < 8; ++k) {
                v16bf aq = *(const v16bf*)(Qb + k*32);
                v16bf b0 = *(const v16bf*)(K0 + k*32);
                v16bf b1 = *(const v16bf*)(K1 + k*32);
                s0 = __builtin_amdgcn_wmma_f32_16x16x32_bf16(false, aq, false, b0, (short)0, s0, false, false);
                s1 = __builtin_amdgcn_wmma_f32_16x16x32_bf16(false, aq, false, b1, (short)0, s1, false, false);
            }
            s0 *= ATT_SCALE; s1 *= ATT_SCALE;

            float tm[8], rs[8], cr[8];
#pragma unroll
            for (int v = 0; v < 8; ++v) tm[v] = fmaxf(s0[v], s1[v]);
#pragma unroll
            for (int off = 1; off < 16; off <<= 1) {
#pragma unroll
                for (int v = 0; v < 8; ++v)
                    tm[v] = fmaxf(tm[v], __shfl_xor(tm[v], off, 32));
            }
#pragma unroll
            for (int v = 0; v < 8; ++v) {
                float nm = fmaxf(m[v], tm[v]);
                cr[v] = __expf(m[v] - nm);
                s0[v] = __expf(s0[v] - nm);
                s1[v] = __expf(s1[v] - nm);
                rs[v] = s0[v] + s1[v];
                m[v]  = nm;
            }
#pragma unroll
            for (int off = 1; off < 16; off <<= 1) {
#pragma unroll
                for (int v = 0; v < 8; ++v)
                    rs[v] += __shfl_xor(rs[v], off, 32);
            }
#pragma unroll
            for (int v = 0; v < 8; ++v) l[v] = l[v] * cr[v] + rs[v];

            // transpose P through LDS into A-operand layout
#pragma unroll
            for (int v = 0; v < 8; ++v) {
                int row = v + 8*hl;
                pl[row][col]      = (bf16_t)s0[v];
                pl[row][16 + col] = (bf16_t)s1[v];
            }
            v8f c8;
#pragma unroll
            for (int v = 0; v < 8; ++v) c8[v] = cr[v];

            v16bf ap = *(const v16bf*)&pl[col][hl*16];
#pragma unroll
            for (int nb = 0; nb < 16; ++nb) {
                v16bf bv = *(const v16bf*)(Vb + ((long)(nb*16 + col)) * Nj + kb + hl*16);
                v8f cacc = pv[nb] * c8;
                pv[nb] = __builtin_amdgcn_wmma_f32_16x16x32_bf16(false, ap, false, bv, (short)0, cacc, false, false);
            }
        }
        float rl[8];
#pragma unroll
        for (int v = 0; v < 8; ++v) rl[v] = 1.f / l[v];
#pragma unroll
        for (int nb = 0; nb < 16; ++nb) {
            int d = nb*16 + col;
#pragma unroll
            for (int v = 0; v < 8; ++v) {
                int row = v + 8*hl;
                float val = pv[nb][v] * rl[v];
                if (j == 0) outl[row][d] = val; else outl[row][d] += val;
            }
        }
    }

    // ---- out projection: o = (combined/3) @ Wo^T ----
#pragma unroll
    for (int nb = 0; nb < 16; ++nb) pv[nb] = vzero;
#pragma unroll
    for (int k = 0; k < 8; ++k) {
        const float* crow = &outl[col][k*32 + hl*16];
        v16bf a;
#pragma unroll
        for (int e = 0; e < 16; ++e) a[e] = (bf16_t)(crow[e] * (1.0f / 3.0f));
#pragma unroll
        for (int nb = 0; nb < 16; ++nb) {
            v16bf bb = *(const v16bf*)(Wo + ((long)i * DIM + nb*16 + col) * DIM + k*32 + hl*16);
            pv[nb] = __builtin_amdgcn_wmma_f32_16x16x32_bf16(false, a, false, bb, (short)0, pv[nb], false, false);
        }
    }

    // ---- bias + residual + layernorm + NCHW store ----
    const float* feat = (i == 0) ? f0 : ((i == 1) ? f1 : f2);
    int HW = Ni;
    float sum[8], sq[8];
#pragma unroll
    for (int v = 0; v < 8; ++v) { sum[v] = 0.f; sq[v] = 0.f; }
#pragma unroll
    for (int nb = 0; nb < 16; ++nb) {
        int d = nb*16 + col;
        float bo = ob[i * DIM + d];
#pragma unroll
        for (int v = 0; v < 8; ++v) {
            int t = qt0 + v + 8*hl;
            float x = pv[nb][v] + bo + feat[((long)(b * DIM + d)) * HW + t];
            pv[nb][v] = x;
            sum[v] += x; sq[v] += x * x;
        }
    }
#pragma unroll
    for (int off = 1; off < 16; off <<= 1) {
#pragma unroll
        for (int v = 0; v < 8; ++v) {
            sum[v] += __shfl_xor(sum[v], off, 32);
            sq[v]  += __shfl_xor(sq[v],  off, 32);
        }
    }
    float mn[8], rstd[8];
#pragma unroll
    for (int v = 0; v < 8; ++v) {
        float mean = sum[v] * (1.0f / DIM);
        float var  = sq[v] * (1.0f / DIM) - mean * mean;
        mn[v] = mean;
        rstd[v] = rsqrtf(var + 1e-5f);
    }
    float* outp = out + (long)c_pref[i] * BATCH * DIM;   // base for scale i
#pragma unroll
    for (int nb = 0; nb < 16; ++nb) {
        int d = nb*16 + col;
        float g  = lng[i * DIM + d];
        float b2 = lnb[i * DIM + d];
#pragma unroll
        for (int v = 0; v < 8; ++v) {
            int t = qt0 + v + 8*hl;
            outp[((long)(b * DIM + d)) * HW + t] = (pv[nb][v] - mn[v]) * rstd[v] * g + b2;
        }
    }
}

// ---------------------------------------------------------------------------
extern "C" void kernel_launch(void* const* d_in, const int* in_sizes, int n_in,
                              void* d_out, int out_size, void* d_ws, size_t ws_size,
                              hipStream_t stream) {
    (void)in_sizes; (void)n_in; (void)out_size; (void)ws_size;
    const float* f0  = (const float*)d_in[0];
    const float* f1  = (const float*)d_in[1];
    const float* f2  = (const float*)d_in[2];
    const float* qw  = (const float*)d_in[3];
    const float* qb  = (const float*)d_in[4];
    const float* kw  = (const float*)d_in[5];
    const float* kb  = (const float*)d_in[6];
    const float* vw  = (const float*)d_in[7];
    const float* vb  = (const float*)d_in[8];
    const float* ow  = (const float*)d_in[9];
    const float* ob  = (const float*)d_in[10];
    const float* lng = (const float*)d_in[11];
    const float* lnb = (const float*)d_in[12];

    char* ws = (char*)d_ws;
    size_t off = 0;
    auto alloc = [&](size_t bytes) -> void* {
        void* p = ws + off;
        off += (bytes + 255) & ~(size_t)255;
        return p;
    };
    const size_t WBYTES = (size_t)NSCALE * DIM * DIM * 2;      // 393216
    const size_t XBYTES = (size_t)BATCH * NTOT * DIM * 2;      // 11010048
    bf16_t* Wq = (bf16_t*)alloc(WBYTES);
    bf16_t* Wk = (bf16_t*)alloc(WBYTES);
    bf16_t* Wv = (bf16_t*)alloc(WBYTES);
    bf16_t* Wo = (bf16_t*)alloc(WBYTES);
    bf16_t* X  = (bf16_t*)alloc(XBYTES);
    bf16_t* Qp = (bf16_t*)alloc(XBYTES);
    bf16_t* Kp = (bf16_t*)alloc(XBYTES);
    bf16_t* Vt = (bf16_t*)alloc(XBYTES);

    const int WN = NSCALE * DIM * DIM;                          // 196608
    cvt_kernel<<<(WN + 255) / 256, 256, 0, stream>>>(qw, Wq, WN);
    cvt_kernel<<<(WN + 255) / 256, 256, 0, stream>>>(kw, Wk, WN);
    cvt_kernel<<<(WN + 255) / 256, 256, 0, stream>>>(vw, Wv, WN);
    cvt_kernel<<<(WN + 255) / 256, 256, 0, stream>>>(ow, Wo, WN);

    pack_kernel<<<(BATCH * NTOT * DIM) / 256, 256, 0, stream>>>(f0, f1, f2, X);

    const int PROJ_BLOCKS = (BATCH * NTOT * 16) / 2 / 16;       // 21504 tiles / 2 waves
    proj_kernel<<<PROJ_BLOCKS, 64, 0, stream>>>(X, Wq, qb, Qp, 0);
    proj_kernel<<<PROJ_BLOCKS, 64, 0, stream>>>(X, Wk, kb, Kp, 0);
    proj_kernel<<<PROJ_BLOCKS, 64, 0, stream>>>(X, Wv, vb, Vt, 1);

    const int ATT_BLOCKS = (1024 + 256 + 64) / 2;               // 1344 tiles / 2 waves
    attn_kernel<<<ATT_BLOCKS, 64, 0, stream>>>(Qp, Kp, Vt, Wo, ob, lng, lnb,
                                               f0, f1, f2, (float*)d_out);
}
